// MyModel_18305150615706
// MI455X (gfx1250) — matrix operand order repaired and checked by reading, not compiled, not run on hardware
//
#include <hip/hip_runtime.h>
#include <cstddef>

// ---------------------------------------------------------------------------
// Fused 3-layer LSTM for MI455X (gfx1250).
// bf16 WMMA (16x16x32) gate GEMMs, all weights LDS-resident (320KB/WGP),
// TDM (tensor_load_to_lds) double-buffered x staging, paired-tile WMMA chains.
// ---------------------------------------------------------------------------

typedef __attribute__((ext_vector_type(16))) __bf16 v16bf;
typedef __attribute__((ext_vector_type(8)))  float  v8f;
typedef __attribute__((ext_vector_type(4)))  unsigned int v4u;
typedef __attribute__((ext_vector_type(4)))  int v4i;
typedef __attribute__((ext_vector_type(8)))  int v8i;

#ifndef ATHENA_USE_TDM
#define ATHENA_USE_TDM 1
#endif
#if ATHENA_USE_TDM && defined(__HIP_DEVICE_COMPILE__) && \
    __has_builtin(__builtin_amdgcn_tensor_load_to_lds) && \
    __has_builtin(__builtin_amdgcn_s_wait_tensorcnt)
#define HAS_TDM 1
#else
#define HAS_TDM 0
#endif

// problem constants
constexpr int Bm = 512, T = 1024, D = 64;
// transposed+padded bf16 weights [Npad][Kpad] packed into one blob
constexpr int OFF_WX0 = 0;                       // [304][64]
constexpr int OFF_WH0 = OFF_WX0 + 304 * 64;      // [304][96]
constexpr int OFF_WX1 = OFF_WH0 + 304 * 96;      // [208][96]
constexpr int OFF_WH1 = OFF_WX1 + 208 * 96;      // [208][64]
constexpr int OFF_WX2 = OFF_WH1 + 208 * 64;      // [112][64]
constexpr int OFF_WH2 = OFF_WX2 + 112 * 64;      // [112][32]
constexpr int WBLOB_HALVES = OFF_WH2 + 112 * 32; // 92672
constexpr int WBLOB_BYTES  = WBLOB_HALVES * 2;   // 185344 (16B multiple)

// LDS layout (16B aligned); total < 320KB per WGP
constexpr int BIAS_OFF = WBLOB_BYTES;            // 624 f32 (304+208+112)
constexpr int XF0_OFF  = BIAS_OFF + 2496;        // 16x64 f32 x tile, buffer 0
constexpr int XF1_OFF  = XF0_OFF + 4096;         // 16x64 f32 x tile, buffer 1
constexpr int H0_OFF   = XF1_OFF + 4096;         // 16x96 bf16 (zero padded)
constexpr int H1_OFF   = H0_OFF + 3072;          // 16x64 bf16
constexpr int H2_OFF   = H1_OFF + 2048;          // 16x32 bf16
constexpr int C0_OFF   = H2_OFF + 1024;          // 16x80 f32
constexpr int C1_OFF   = C0_OFF + 5120;          // 16x64 f32
constexpr int C2_OFF   = C1_OFF + 4096;          // 16x32 f32
constexpr int G_OFF    = C2_OFF + 2048;          // 16x304 f32 gates
constexpr int LDS_BYTES = G_OFF + 19456;         // 232896

__device__ __forceinline__ unsigned short f2bf(float f) {
    unsigned int u = __float_as_uint(f);
    return (unsigned short)((u + 0x7fffu + ((u >> 16) & 1u)) >> 16);  // RNE
}
__device__ __forceinline__ float bf2f(unsigned short h) {
    return __uint_as_float(((unsigned int)h) << 16);
}
__device__ __forceinline__ float sigm(float x) { return 1.0f / (1.0f + __expf(-x)); }

// bf16 16x32 A fragment (ISA 7.12.2): lanes0-15 M=lane, K=0..7 & 16..23;
// lanes16-31 M=lane-16, K=8..15 & 24..31. p = row base + ka.
__device__ __forceinline__ v16bf load_a_frag(const unsigned short* p) {
    const __bf16* q = (const __bf16*)__builtin_assume_aligned(p, 16);
    v16bf r;
#pragma unroll
    for (int i = 0; i < 8; ++i) { r[i] = q[i]; r[i + 8] = q[i + 16]; }
    return r;
}
// same fragment but sourced from f32 LDS (x tile), converting on the fly
__device__ __forceinline__ v16bf load_a_frag_f32(const float* p) {
    const float* q = (const float*)__builtin_assume_aligned(p, 16);
    v16bf r;
#pragma unroll
    for (int i = 0; i < 8; ++i) { r[i] = (__bf16)q[i]; r[i + 8] = (__bf16)q[i + 16]; }
    return r;
}
// bf16 32x16 B fragment: lane N=lane&15 holds K=kb..kb+7 & kb+16..kb+23
__device__ __forceinline__ v16bf load_b_frag(const unsigned short* p) {
    const __bf16* q = (const __bf16*)__builtin_assume_aligned(p, 16);
    v16bf r;
#pragma unroll
    for (int i = 0; i < 8; ++i) { r[i] = q[i]; r[i + 8] = q[i + 16]; }
    return r;
}

#define WMMA_BF16(Afrag, Bfrag, Cacc) \
    __builtin_amdgcn_wmma_f32_16x16x32_bf16(false, (Afrag), false, (Bfrag), (short)0, (Cacc), false, false)

// gate GEMM: g[16][4H'] = A_x*Wx + A_h*Wh + bias. N-tiles strided over 8 waves.
// Tiles are processed in PAIRS with split X/H accumulators: up to 4 independent
// WMMA chains so the XDL pipe never stalls on D->C dependencies.
template <int KSX, int KSH>
__device__ __forceinline__ void wave_gemm(
    const v16bf* __restrict__ ax, const v16bf* __restrict__ ah,
    const unsigned short* __restrict__ wX, int KpX,
    const unsigned short* __restrict__ wH, int KpH,
    const float* __restrict__ bias, float* __restrict__ g, int ldg,
    int ntiles, int wave, int lane)
{
    const int  nn = lane & 15;
    const bool hi = lane >= 16;
    const int  kb = hi ? 8 : 0;
    const int  r0 = hi ? 8 : 0;

    int tile = wave;
    for (; tile + 8 < ntiles; tile += 16) {
        const int nA = tile * 16 + nn, nB = (tile + 8) * 16 + nn;
        v8f x0, h0, x1, h1;
        const float b0 = bias[nA], b1 = bias[nB];
#pragma unroll
        for (int r = 0; r < 8; ++r) { x0[r] = b0; x1[r] = b1; h0[r] = 0.f; h1[r] = 0.f; }
        const unsigned short* wx0 = wX + (size_t)nA * KpX + kb;
        const unsigned short* wx1 = wX + (size_t)nB * KpX + kb;
        const unsigned short* wh0 = wH + (size_t)nA * KpH + kb;
        const unsigned short* wh1 = wH + (size_t)nB * KpH + kb;
#pragma unroll
        for (int s = 0; s < KSX; ++s) {
            x0 = WMMA_BF16(ax[s], load_b_frag(wx0 + s * 32), x0);
            x1 = WMMA_BF16(ax[s], load_b_frag(wx1 + s * 32), x1);
        }
#pragma unroll
        for (int s = 0; s < KSH; ++s) {
            h0 = WMMA_BF16(ah[s], load_b_frag(wh0 + s * 32), h0);
            h1 = WMMA_BF16(ah[s], load_b_frag(wh1 + s * 32), h1);
        }
#pragma unroll
        for (int r = 0; r < 8; ++r) {
            g[(size_t)(r0 + r) * ldg + nA] = x0[r] + h0[r];
            g[(size_t)(r0 + r) * ldg + nB] = x1[r] + h1[r];
        }
    }
    if (tile < ntiles) {
        const int nA = tile * 16 + nn;
        v8f x0, h0;
        const float b0 = bias[nA];
#pragma unroll
        for (int r = 0; r < 8; ++r) { x0[r] = b0; h0[r] = 0.f; }
        const unsigned short* wx0 = wX + (size_t)nA * KpX + kb;
        const unsigned short* wh0 = wH + (size_t)nA * KpH + kb;
#pragma unroll
        for (int s = 0; s < KSX; ++s)
            x0 = WMMA_BF16(ax[s], load_b_frag(wx0 + s * 32), x0);
#pragma unroll
        for (int s = 0; s < KSH; ++s)
            h0 = WMMA_BF16(ah[s], load_b_frag(wh0 + s * 32), h0);
#pragma unroll
        for (int r = 0; r < 8; ++r)
            g[(size_t)(r0 + r) * ldg + nA] = x0[r] + h0[r];
    }
}

__device__ __forceinline__ void cell_update(
    const float* __restrict__ g, int ldg, int H,
    float* __restrict__ c, int ldc,
    unsigned short* __restrict__ h, int ldh, int tid)
{
    for (int idx = tid; idx < 16 * H; idx += 256) {
        const int m = idx / H;
        const int j = idx - m * H;
        const float* gr = g + (size_t)m * ldg;
        const float iv = sigm(gr[j]);
        const float fv = sigm(gr[H + j]);
        const float cc = tanhf(gr[2 * H + j]);
        const float ov = sigm(gr[3 * H + j]);
        const float cv = fv * c[m * ldc + j] + iv * cc;
        c[m * ldc + j] = cv;
        h[m * ldh + j] = f2bf(ov * tanhf(cv));
    }
}

#if HAS_TDM
// TDM D# (ISA 8.3/8.4): 2D tile 64x16 f32, row stride T*D elems, dense into LDS
__device__ __forceinline__ void tdm_load_x_tile(const float* gsrc, unsigned lds_off)
{
    const unsigned long long ga = (unsigned long long)(size_t)gsrc;
    v4u g0;
    g0[0] = 1u;                                            // count=1, user mode
    g0[1] = lds_off;                                       // lds_addr [63:32]
    g0[2] = (unsigned)(ga & 0xffffffffu);                  // global_addr lo
    g0[3] = (unsigned)((ga >> 32) & 0x01ffffffu) | (2u << 30);  // ga hi | type=2
    v8i g1;
    g1[0] = 0x20000;                  // wg_mask=0, data_size=2 (4B)
    g1[1] = (int)(64u << 16);         // tensor_dim0 = 64
    g1[2] = (int)(16u << 16);         // tensor_dim1 = 16
    g1[3] = (int)(64u << 16);         // tile_dim0 = 64
    g1[4] = 16;                       // tile_dim1 = 16, tile_dim2 = 0
    g1[5] = (int)(T * D);             // tensor_dim0_stride = 65536 elems
    g1[6] = 0;
    g1[7] = 0;
    v4i z4 = {0, 0, 0, 0};
#if __clang_major__ >= 23
    v8i z8 = {0, 0, 0, 0, 0, 0, 0, 0};
    __builtin_amdgcn_tensor_load_to_lds(g0, g1, z4, z4, z8, 0);
#else
    __builtin_amdgcn_tensor_load_to_lds(g0, g1, z4, z4, 0);
#endif
}
#endif

__global__ __launch_bounds__(256, 1) void lstm3_fused(
    const float* __restrict__ x,
    const unsigned short* __restrict__ wblob,
    const float* __restrict__ bpad,
    const float* __restrict__ Wd,
    const float* __restrict__ bd,
    float* __restrict__ out)
{
    extern __shared__ char smem[];
    unsigned short* sW  = (unsigned short*)smem;
    float*          sB  = (float*)(smem + BIAS_OFF);
    unsigned short* sh0 = (unsigned short*)(smem + H0_OFF);
    unsigned short* sh1 = (unsigned short*)(smem + H1_OFF);
    unsigned short* sh2 = (unsigned short*)(smem + H2_OFF);
    float*          c0  = (float*)(smem + C0_OFF);
    float*          c1  = (float*)(smem + C1_OFF);
    float*          c2  = (float*)(smem + C2_OFF);
    float*          g   = (float*)(smem + G_OFF);

    const int tid  = threadIdx.x;
    const int lane = tid & 31;
    const int wave = tid >> 5;
    const int m0   = blockIdx.x * 16;

    const int  row = lane & 15;
    const bool hi  = lane >= 16;
    const int  ka  = hi ? 8 : 0;

    // one-time: weights + biases -> LDS, zero h/c state
    {
        const uint4* s4 = (const uint4*)wblob;
        uint4*       d4w = (uint4*)sW;
        for (int i = tid; i < WBLOB_BYTES / 16; i += 256) d4w[i] = s4[i];
        for (int i = tid; i < 624; i += 256) sB[i] = bpad[i];
        uint4 z; z.x = z.y = z.z = z.w = 0u;
        uint4* zp = (uint4*)(smem + H0_OFF);
        for (int i = tid; i < (G_OFF - H0_OFF) / 16; i += 256) zp[i] = z;
    }

    const float* xtile = x + (size_t)m0 * T * D;   // this block's batch tile
#if HAS_TDM
    if (wave == 0) tdm_load_x_tile(xtile, XF0_OFF);     // prime t=0
#endif
    __syncthreads();

    int cur = 0;
    for (int t = 0; t < T; ++t) {
        const float* xcur = (const float*)(smem + (cur ? XF1_OFF : XF0_OFF));
#if HAS_TDM
        if (wave == 0) __builtin_amdgcn_s_wait_tensorcnt(0);
#else
        {   // fallback: cooperative synchronous stage into current buffer
            const int m  = tid >> 4;
            const int d4 = (tid & 15) * 4;
            const float* xp = xtile + ((size_t)m * T + t) * 1 * D + d4;
            float4 xv = *(const float4*)xp;
            *(float4*)((float*)xcur + m * D + d4) = xv;
            if (t + 1 < T) __builtin_prefetch(xp + D, 0, 3);
        }
#endif
        __syncthreads();
#if HAS_TDM
        if (wave == 0 && t + 1 < T)
            tdm_load_x_tile(xtile + (size_t)(t + 1) * D, cur ? XF0_OFF : XF1_OFF);
#endif

        // ---- layer 0: g[16x300] = x(16x64)Wx0 + h0(16x96)Wh0 + b0 (19 tiles)
        {
            v16bf ax[2], ah[3];
#pragma unroll
            for (int s = 0; s < 2; ++s) ax[s] = load_a_frag_f32(xcur + row * 64 + s * 32 + ka);
#pragma unroll
            for (int s = 0; s < 3; ++s) ah[s] = load_a_frag(sh0 + row * 96 + s * 32 + ka);
            wave_gemm<2, 3>(ax, ah, sW + OFF_WX0, 64, sW + OFF_WH0, 96,
                            sB + 0, g, 304, 19, wave, lane);
        }
        __syncthreads();
        cell_update(g, 304, 75, c0, 80, sh0, 96, tid);
        __syncthreads();

        // ---- layer 1: g[16x200] = h0(16x96)Wx1 + h1(16x64)Wh1 + b1 (13 tiles)
        {
            v16bf ax[3], ah[2];
#pragma unroll
            for (int s = 0; s < 3; ++s) ax[s] = load_a_frag(sh0 + row * 96 + s * 32 + ka);
#pragma unroll
            for (int s = 0; s < 2; ++s) ah[s] = load_a_frag(sh1 + row * 64 + s * 32 + ka);
            wave_gemm<3, 2>(ax, ah, sW + OFF_WX1, 96, sW + OFF_WH1, 64,
                            sB + 304, g, 208, 13, wave, lane);
        }
        __syncthreads();
        cell_update(g, 208, 50, c1, 64, sh1, 64, tid);
        __syncthreads();

        // ---- layer 2: g[16x100] = h1(16x64)Wx2 + h2(16x32)Wh2 + b2 (7 tiles)
        {
            v16bf ax[2], ah[1];
#pragma unroll
            for (int s = 0; s < 2; ++s) ax[s] = load_a_frag(sh1 + row * 64 + s * 32 + ka);
            ah[0] = load_a_frag(sh2 + row * 32 + ka);
            wave_gemm<2, 1>(ax, ah, sW + OFF_WX2, 64, sW + OFF_WH2, 32,
                            sB + 512, g, 112, 7, wave, lane);
        }
        __syncthreads();
        cell_update(g, 112, 25, c2, 32, sh2, 32, tid);
        __syncthreads();

        cur ^= 1;
    }

    // dense head on last h2: out = sigmoid(h2 @ Wd + bd)
    if (tid < 16) {
        float s = bd[0];
#pragma unroll
        for (int j = 0; j < 25; ++j) s += bf2f(sh2[tid * 32 + j]) * Wd[j];
        out[m0 + tid] = sigm(s);
    }
}

// ---- prep: fp32 [K][N] weight -> bf16 transposed+padded [Np][Kp] in ws ----
__global__ void prep_wt(const float* __restrict__ W, int K, int N, int Kp, int Np,
                        unsigned short* __restrict__ outp)
{
    int i = blockIdx.x * blockDim.x + threadIdx.x;
    if (i >= Np * Kp) return;
    int n = i / Kp, k = i - n * Kp;
    float v = (k < K && n < N) ? W[(size_t)k * N + n] : 0.0f;
    outp[i] = f2bf(v);
}
__global__ void prep_bias(const float* __restrict__ b, int N, int Np,
                          float* __restrict__ outp)
{
    int i = blockIdx.x * blockDim.x + threadIdx.x;
    if (i < Np) outp[i] = (i < N) ? b[i] : 0.0f;
}

extern "C" void kernel_launch(void* const* d_in, const int* in_sizes, int n_in,
                              void* d_out, int out_size, void* d_ws, size_t ws_size,
                              hipStream_t stream)
{
    (void)in_sizes; (void)n_in; (void)out_size; (void)ws_size;
    const float* x   = (const float*)d_in[0];
    const float* Wx0 = (const float*)d_in[1];
    const float* Wh0 = (const float*)d_in[2];
    const float* b0  = (const float*)d_in[3];
    const float* Wx1 = (const float*)d_in[4];
    const float* Wh1 = (const float*)d_in[5];
    const float* b1  = (const float*)d_in[6];
    const float* Wx2 = (const float*)d_in[7];
    const float* Wh2 = (const float*)d_in[8];
    const float* b2  = (const float*)d_in[9];
    const float* Wd  = (const float*)d_in[10];
    const float* bd  = (const float*)d_in[11];

    unsigned short* wblob = (unsigned short*)d_ws;
    float*          bpad  = (float*)((char*)d_ws + WBLOB_BYTES);

    auto wt = [&](const float* W, int K, int N, int Kp, int Np, int off) {
        int n = Np * Kp;
        prep_wt<<<(n + 255) / 256, 256, 0, stream>>>(W, K, N, Kp, Np, wblob + off);
    };
    wt(Wx0, 64, 300, 64, 304, OFF_WX0);
    wt(Wh0, 75, 300, 96, 304, OFF_WH0);
    wt(Wx1, 75, 200, 96, 208, OFF_WX1);
    wt(Wh1, 50, 200, 64, 208, OFF_WH1);
    wt(Wx2, 50, 100, 64, 112, OFF_WX2);
    wt(Wh2, 25, 100, 32, 112, OFF_WH2);
    prep_bias<<<2, 256, 0, stream>>>(b0, 300, 304, bpad + 0);
    prep_bias<<<1, 256, 0, stream>>>(b1, 200, 208, bpad + 304);
    prep_bias<<<1, 128, 0, stream>>>(b2, 100, 112, bpad + 512);

    (void)hipFuncSetAttribute((const void*)lstm3_fused,
                              hipFuncAttributeMaxDynamicSharedMemorySize, LDS_BYTES);

    lstm3_fused<<<Bm / 16, 256, LDS_BYTES, stream>>>(x, wblob, bpad, Wd, bd,
                                                     (float*)d_out);
}
